// ECGRGNN_1211180778320
// MI455X (gfx1250) — matrix operand order repaired and checked by reading, not compiled, not run on hardware
//
#include <hip/hip_runtime.h>
#include <hip/hip_bf16.h>

typedef _Float16 h16v __attribute__((ext_vector_type(16)));
typedef _Float16 h8v  __attribute__((ext_vector_type(8)));
typedef float    f8v  __attribute__((ext_vector_type(8)));

constexpr int NG = 512;   // graphs
constexpr int NV = 6144;  // nodes (512*12)
constexpr int T  = 500;
constexpr int H  = 64;
constexpr int G  = 128;
constexpr int NC = 5;

// ---------------------------------------------------------------- helpers ---
__device__ __forceinline__ f8v zero8() {
  f8v z;
#pragma unroll
  for (int i = 0; i < 8; ++i) z[i] = 0.0f;
  return z;
}

__device__ __forceinline__ f8v wmma_f16(h16v a, h16v b, f8v c) {
  // v_wmma_f32_16x16x32_f16: (neg_a, A, neg_b, B, c_mod, C, reuse_a, reuse_b)
  return __builtin_amdgcn_wmma_f32_16x16x32_f16(false, a, false, b, (short)0, c,
                                                false, false);
}

// Raw hardware transcendentals: avoid libm's IEEE div / branchy tanh.
__device__ __forceinline__ float rcp_hw(float x) {
#if __has_builtin(__builtin_amdgcn_rcpf)
  return __builtin_amdgcn_rcpf(x);        // v_rcp_f32
#else
  return 1.0f / x;
#endif
}

__device__ __forceinline__ float exp2_hw(float x) {
#if __has_builtin(__builtin_amdgcn_exp2f)
  return __builtin_amdgcn_exp2f(x);       // v_exp_f32
#else
  return __expf(x * 0.6931471805599453f);
#endif
}

__device__ __forceinline__ float rsq_hw(float x) {
#if __has_builtin(__builtin_amdgcn_rsqf)
  return __builtin_amdgcn_rsqf(x);        // v_rsq_f32
#else
  return rsqrtf(x);
#endif
}

__device__ __forceinline__ float sigmoid_fast(float x) {
  // 1 / (1 + exp(-x)) = 1 / (1 + exp2(-x*log2e))
  float e = exp2_hw(x * -1.4426950408889634f);
  return rcp_hw(1.0f + e);
}

__device__ __forceinline__ float tanh_fast(float x) {
#if __has_builtin(__builtin_amdgcn_tanhf)
  return __builtin_amdgcn_tanhf(x);       // gfx1250 v_tanh_f32
#else
  // tanh(x) = 1 - 2/(exp(2x)+1); saturates to +/-1 for large |x|.
  float e = exp2_hw(x * 2.8853900817779268f);  // exp(2x)
  return 1.0f - 2.0f * rcp_hw(e + 1.0f);
#endif
}

// A fragment (16x32 f16), row-major tile with row stride ldk halves.
// Lane layout (ISA 7.12.2): lanes 0-15 K=[kb..kb+7],[kb+16..kb+23];
// lanes 16-31 K=[kb+8..kb+15],[kb+24..kb+31].
__device__ __forceinline__ h16v frag_A(const _Float16* base, int lane, int ldk,
                                       int kBase) {
  int m  = lane & 15;
  int ks = (lane >> 4) * 8;
  const _Float16* p = base + m * ldk + kBase + ks;
  h8v lo = *(const h8v*)(p);
  h8v hi = *(const h8v*)(p + 16);
  h16v f;
#pragma unroll
  for (int i = 0; i < 8; ++i) { f[i] = lo[i]; f[i + 8] = hi[i]; }
  return f;
}

// B fragment (32x16 f16) from transposed source BT[n][k], row stride ldk halves.
// Lanes 0-15 hold K=[0..7],[16..23] of column n, lanes 16-31 hold K=[8..15],[24..31].
__device__ __forceinline__ h16v frag_B(const _Float16* baseT, int lane, int ldk,
                                       int kBase) {
  int n  = lane & 15;
  int ks = (lane >> 4) * 8;
  const _Float16* p = baseT + n * ldk + kBase + ks;
  h8v lo = *(const h8v*)(p);
  h8v hi = *(const h8v*)(p + 16);
  h16v f;
#pragma unroll
  for (int i = 0; i < 8; ++i) { f[i] = lo[i]; f[i + 8] = hi[i]; }
  return f;
}

// ----------------------------------------------------------- transpose x ----
__global__ __launch_bounds__(256) void transpose_k(const float* __restrict__ x,
                                                   float* __restrict__ xT) {
  int idx = blockIdx.x * 256 + threadIdx.x;
  if (idx >= NV * T) return;
  int n = idx / T;
  int t = idx - n * T;           // coalesced read along t
  xT[(size_t)t * NV + n] = x[idx];
}

// ------------------------------------------------------------- GRU scan -----
// Block = 128 threads = 4 waves; each wave owns 16 rows. Grid = NV/64 = 96.
// LDS: wT[192][96]  augmented transposed recurrence weights (36 KB)
//      wig[64][32]  input-gate weights for the 'n' columns (4 KB)
//      hA [64][96]  augmented hidden state [h | x_t | 1 | 0] as f16 (12 KB)
// Hidden state is ALSO kept in f32 registers (D-layout) across steps so the
// gate blend never re-reads LDS; the LDS f16 copy only feeds A-fragments.
__global__ __launch_bounds__(128) void gru_k(const float* __restrict__ xT,
                                             const float* __restrict__ w_hh,
                                             const float* __restrict__ w_ih,
                                             const float* __restrict__ b_ih,
                                             const float* __restrict__ b_hh,
                                             float* __restrict__ Hout) {
  __shared__ _Float16 wT[192 * 96];
  __shared__ _Float16 wig[64 * 32];
  __shared__ _Float16 hA[64 * 96];

  const int tid  = threadIdx.x;
  const int lane = tid & 31;
  const int wv   = tid >> 5;
  const int row0 = blockIdx.x * 64;

  // Build augmented weights: wT[c][k]; c = output gate column (0..191).
  for (int i = tid; i < 192 * 96; i += 128) {
    int c = i / 96, k = i - c * 96;
    float v = 0.0f;
    if (k < 64)       v = w_hh[c * 64 + k];                          // h part
    else if (k == 64) v = (c < 128) ? w_ih[c] : 0.0f;                // x row
    else if (k == 65) v = (c < 128) ? (b_ih[c] + b_hh[c]) : b_hh[c]; // 1 row
    wT[i] = (_Float16)v;
  }
  // ig = x*w_in + b_ih for the 'n' gate columns (global cols 128..191)
  for (int i = tid; i < 64 * 32; i += 128) {
    int n = i >> 5, k = i & 31;
    float v = (k == 0) ? w_ih[128 + n] : ((k == 1) ? b_ih[128 + n] : 0.0f);
    wig[i] = (_Float16)v;
  }
  // h0 = 0, ones column at k=65
  for (int i = tid; i < 64 * 96; i += 128) {
    int k = i % 96;
    hA[i] = (_Float16)((k == 65) ? 1.0f : 0.0f);
  }
  __syncthreads();

  _Float16* myH    = hA + wv * (16 * 96);
  const int myRow0 = row0 + wv * 16;

  // Persistent f32 hidden state in the C/D register layout:
  // h_state[g][i] = h[row rb+i][col g*16 + (lane&15)]
  f8v h_state[4];
#pragma unroll
  for (int g = 0; g < 4; ++g) h_state[g] = zero8();

  for (int t = 0; t < T; ++t) {
    if (lane < 16) {
      myH[lane * 96 + 64] = (_Float16)xT[(size_t)t * NV + myRow0 + lane];
      if (t + 1 < T)
        __builtin_prefetch(&xT[(size_t)(t + 1) * NV + myRow0 + lane], 0, 0);
    }
    __syncthreads();  // order previous-step h stores vs this step's A loads

    h16v a0 = frag_A(myH, lane, 96, 0);
    h16v a1 = frag_A(myH, lane, 96, 32);
    h16v a2 = frag_A(myH, lane, 96, 64);

#pragma unroll
    for (int g = 0; g < 4; ++g) {
      const _Float16* bR = wT + ((0 + g) * 16) * 96;   // cols g*16      (r)
      const _Float16* bZ = wT + ((4 + g) * 16) * 96;   // cols 64+g*16   (z)
      const _Float16* bN = wT + ((8 + g) * 16) * 96;   // cols 128+g*16  (hg)

      f8v ar = zero8(), az = zero8(), ah = zero8();
      ar = wmma_f16(a0, frag_B(bR, lane, 96, 0),  ar);
      ar = wmma_f16(a1, frag_B(bR, lane, 96, 32), ar);
      ar = wmma_f16(a2, frag_B(bR, lane, 96, 64), ar);
      az = wmma_f16(a0, frag_B(bZ, lane, 96, 0),  az);
      az = wmma_f16(a1, frag_B(bZ, lane, 96, 32), az);
      az = wmma_f16(a2, frag_B(bZ, lane, 96, 64), az);
      ah = wmma_f16(a0, frag_B(bN, lane, 96, 0),  ah);
      ah = wmma_f16(a1, frag_B(bN, lane, 96, 32), ah);
      ah = wmma_f16(a2, frag_B(bN, lane, 96, 64), ah);
      f8v ai = wmma_f16(a2, frag_B(wig + (g * 16) * 32, lane, 32, 0), zero8());

      // Gate math in the C/D layout: element i -> row rb+i, column col.
      const int col = g * 16 + (lane & 15);
      const int rb  = (lane >> 4) * 8;
#pragma unroll
      for (int i = 0; i < 8; ++i) {
        float r  = sigmoid_fast(ar[i]);
        float z  = sigmoid_fast(az[i]);
        float nn = tanh_fast(ai[i] + r * ah[i]);
        float hn = (1.0f - z) * nn + z * h_state[g][i];
        h_state[g][i] = hn;
        myH[(rb + i) * 96 + col] = (_Float16)hn;   // f16 copy for A-fragments
      }
    }
  }
  __syncthreads();

  // Write final hidden state (f32, register-resident) to global.
  {
    const int col = lane & 15;
    const int rb  = (lane >> 4) * 8;
#pragma unroll
    for (int g = 0; g < 4; ++g)
#pragma unroll
      for (int i = 0; i < 8; ++i)
        Hout[(size_t)(myRow0 + rb + i) * H + g * 16 + col] = h_state[g][i];
  }
}

// ------------------------------------------------------- generic WMMA GEMM --
// C[M x Nc] = A[M x K] * B[K x Nc] (+ bias). One wave per 16x16 C tile.
// Grid: (M/16, ceil(Nc/16)). K in {64,128}, multiple of 32.
__global__ __launch_bounds__(32) void wmma_gemm_k(const float* __restrict__ A,
                                                  const float* __restrict__ B,
                                                  const float* __restrict__ bias,
                                                  float* __restrict__ C,
                                                  int K, int Nc, int addBias) {
  __shared__ _Float16 sA[16 * 128];
  __shared__ _Float16 sBT[16 * 128];
  const int lane = threadIdx.x;
  const int m0 = blockIdx.x * 16;
  const int n0 = blockIdx.y * 16;

  for (int i = lane; i < 16 * K; i += 32) {
    int r = i / K, k = i - r * K;
    sA[r * K + k] = (_Float16)A[(size_t)(m0 + r) * K + k];
  }
  for (int i = lane; i < 16 * K; i += 32) {
    int n = i / K, k = i - n * K;
    float v = (n0 + n < Nc) ? B[(size_t)k * Nc + (n0 + n)] : 0.0f;
    sBT[n * K + k] = (_Float16)v;
  }
  __syncthreads();

  f8v acc = zero8();
  for (int kc = 0; kc < K; kc += 32)
    acc = wmma_f16(frag_A(sA, lane, K, kc), frag_B(sBT, lane, K, kc), acc);

  const int col = n0 + (lane & 15);
  const int rb  = (lane >> 4) * 8;
  if (col < Nc) {
#pragma unroll
    for (int i = 0; i < 8; ++i) {
      float v = acc[i] + (addBias ? bias[col] : 0.0f);
      C[(size_t)(m0 + rb + i) * Nc + col] = v;
    }
  }
}

// ----------------------------------------------------------- GCN pieces -----
__global__ __launch_bounds__(256) void zero_k(float* __restrict__ p, int n) {
  int i = blockIdx.x * 256 + threadIdx.x;
  if (i < n) p[i] = 0.0f;
}

__global__ __launch_bounds__(256) void deg_k(const int* __restrict__ dst,
                                             float* __restrict__ deg, int E) {
  int e = blockIdx.x * 256 + threadIdx.x;
  if (e < E) atomicAdd(&deg[dst[e]], 1.0f);
}

__global__ __launch_bounds__(256) void edge_agg_k(const int* __restrict__ src,
                                                  const int* __restrict__ dst,
                                                  const float* __restrict__ deg,
                                                  const float* __restrict__ xw,
                                                  float* __restrict__ out, int E) {
  int idx = blockIdx.x * 256 + threadIdx.x;
  if (idx >= E * G) return;
  int e = idx / G, f = idx - e * G;
  int s = src[e], d = dst[e];
  float norm = rsq_hw(deg[s] + 1.0f) * rsq_hw(deg[d] + 1.0f);
  atomicAdd(&out[(size_t)d * G + f], xw[(size_t)s * G + f] * norm);
}

__global__ __launch_bounds__(256) void self_bias_relu_k(const float* __restrict__ xw,
                                                        const float* __restrict__ deg,
                                                        const float* __restrict__ b,
                                                        float* __restrict__ out) {
  int idx = blockIdx.x * 256 + threadIdx.x;
  if (idx >= NV * G) return;
  int i = idx / G, f = idx - i * G;
  float dinv2 = rcp_hw(deg[i] + 1.0f);
  float v = out[idx] + xw[idx] * dinv2 + b[f];
  out[idx] = fmaxf(v, 0.0f);
}

__global__ __launch_bounds__(256) void pool_k(const float* __restrict__ x,
                                              float* __restrict__ pooled) {
  int idx = blockIdx.x * 256 + threadIdx.x;
  if (idx >= NG * G) return;
  int g = idx / G, f = idx - g * G;
  float s = 0.0f;
#pragma unroll
  for (int l = 0; l < 12; ++l) s += x[(size_t)(g * 12 + l) * G + f];
  pooled[idx] = s * (1.0f / 12.0f);
}

// ------------------------------------------------------------- launcher -----
extern "C" void kernel_launch(void* const* d_in, const int* in_sizes, int n_in,
                              void* d_out, int out_size, void* d_ws, size_t ws_size,
                              hipStream_t stream) {
  const float* x_raw = (const float*)d_in[0];
  const int*   ei    = (const int*)d_in[1];
  // d_in[2] = batch (structure is fixed: node i -> graph i/12)
  const float* w_ih = (const float*)d_in[3];
  const float* w_hh = (const float*)d_in[4];
  const float* b_ih = (const float*)d_in[5];
  const float* b_hh = (const float*)d_in[6];
  const float* W1   = (const float*)d_in[7];
  const float* b1   = (const float*)d_in[8];
  const float* W2   = (const float*)d_in[9];
  const float* b2   = (const float*)d_in[10];
  const float* Wfc  = (const float*)d_in[11];
  const float* bfc  = (const float*)d_in[12];
  const int E = in_sizes[1] / 2;
  const int* src = ei;
  const int* dst = ei + E;

  // workspace layout (floats); peak ~20.4 MB
  float* ws     = (float*)d_ws;
  float* xT     = ws;                          // T*NV
  float* Hbuf   = xT   + (size_t)T * NV;       // NV*H
  float* buf1   = Hbuf + (size_t)NV * H;       // NV*G
  float* buf2   = buf1 + (size_t)NV * G;       // NV*G
  float* deg    = buf2 + (size_t)NV * G;       // NV
  float* pooled = deg  + NV;                   // NG*G

  // 1. transpose x for coalesced per-step access
  transpose_k<<<(NV * T + 255) / 256, 256, 0, stream>>>(x_raw, xT);

  // 2. GRU scan (WMMA-based recurrence)
  gru_k<<<NV / 64, 128, 0, stream>>>(xT, w_hh, w_ih, b_ih, b_hh, Hbuf);

  // degrees (shared by both GCN layers)
  zero_k<<<(NV + 255) / 256, 256, 0, stream>>>(deg, NV);
  deg_k<<<(E + 255) / 256, 256, 0, stream>>>(dst, deg, E);

  // 3. GCN layer 1: XW1 = H @ W1 -> buf1 ; aggregate -> buf2 (=A1)
  wmma_gemm_k<<<dim3(NV / 16, G / 16), 32, 0, stream>>>(Hbuf, W1, nullptr, buf1,
                                                        H, G, 0);
  zero_k<<<(NV * G + 255) / 256, 256, 0, stream>>>(buf2, NV * G);
  edge_agg_k<<<(E * G + 255) / 256, 256, 0, stream>>>(src, dst, deg, buf1, buf2, E);
  self_bias_relu_k<<<(NV * G + 255) / 256, 256, 0, stream>>>(buf1, deg, b1, buf2);

  // 4. GCN layer 2: XW2 = A1 @ W2 -> buf1 ; aggregate -> buf2 (=A2)
  wmma_gemm_k<<<dim3(NV / 16, G / 16), 32, 0, stream>>>(buf2, W2, nullptr, buf1,
                                                        G, G, 0);
  zero_k<<<(NV * G + 255) / 256, 256, 0, stream>>>(buf2, NV * G);
  edge_agg_k<<<(E * G + 255) / 256, 256, 0, stream>>>(src, dst, deg, buf1, buf2, E);
  self_bias_relu_k<<<(NV * G + 255) / 256, 256, 0, stream>>>(buf1, deg, b2, buf2);

  // 5. mean pool per graph
  pool_k<<<(NG * G + 255) / 256, 256, 0, stream>>>(buf2, pooled);

  // 6. final FC: out = pooled @ Wfc + bfc  (Nc=5, padded to one 16-col tile)
  wmma_gemm_k<<<dim3(NG / 16, 1), 32, 0, stream>>>(pooled, Wfc, bfc, (float*)d_out,
                                                   G, NC, 1);
}
